// _S6Layer_20040317403360
// MI455X (gfx1250) — matrix operand order repaired and checked by reading, not compiled, not run on hardware
//
#include <hip/hip_runtime.h>
#include <math.h>

#define D_MODEL 1024
#define D_STATE 16
#define D_CONV  4
#define D_INNER 2048
#define SEQ_B   2
#define SEQ_T   2048
#define NTOK    (SEQ_B * SEQ_T)          /* 4096 tokens */
#define NBCDT   (2 * D_STATE + 1)        /* 33 */
#define LN_EPS  1e-5f

typedef __attribute__((ext_vector_type(16))) __bf16 v16bf;
typedef __attribute__((ext_vector_type(8)))  float  v8f;
typedef int v4i __attribute__((vector_size(16)));   // matches builtin param type

union FragAB { v16bf v; unsigned int u[8]; };
union FragC  { v8f   v; float        f[8]; };

__device__ __forceinline__ unsigned short f2bf(float x) {
  union { float f; unsigned int u; } c; c.f = x;
  unsigned int u = c.u;
  u += 0x7fffu + ((u >> 16) & 1u);     // round-to-nearest-even
  return (unsigned short)(u >> 16);
}

// ---------------------------------------------------------------------------
// Async global->LDS (gfx1250) with safe fallback
// ---------------------------------------------------------------------------
#if defined(__has_builtin)
#if __has_builtin(__builtin_amdgcn_global_load_async_to_lds_b128)
#define HAVE_ASYNC_LDS 1
#endif
#endif
#ifndef HAVE_ASYNC_LDS
#define HAVE_ASYNC_LDS 0
#endif

#define AS1 __attribute__((address_space(1)))
#define AS3 __attribute__((address_space(3)))

__device__ __forceinline__ void cp16(void* lds, const void* g) {
#if HAVE_ASYNC_LDS
  __builtin_amdgcn_global_load_async_to_lds_b128(
      (AS1 v4i*)(unsigned long long)(g),
      (AS3 v4i*)(unsigned int)(unsigned long long)(lds), 0, 0);
#else
  *(uint4*)lds = *(const uint4*)g;
#endif
}

__device__ __forceinline__ void cp_wait() {
#if HAVE_ASYNC_LDS
#if defined(__has_builtin) && __has_builtin(__builtin_amdgcn_s_wait_asynccnt)
  __builtin_amdgcn_s_wait_asynccnt(0);
#else
  asm volatile("s_wait_asynccnt 0" ::: "memory");
#endif
#endif
}

// ---------------------------------------------------------------------------
// fp32 -> bf16 bulk convert (8 elements / thread, b128 in / b128 out)
// ---------------------------------------------------------------------------
__global__ __launch_bounds__(256)
void convert_bf16_kernel(const float* __restrict__ s,
                         unsigned short* __restrict__ d, int n8)
{
  int i = blockIdx.x * 256 + threadIdx.x;
  if (i >= n8) return;
  const float4* s4 = (const float4*)s + (size_t)i * 2;
  float4 f0 = s4[0], f1 = s4[1];
  uint4 o;
  o.x = (unsigned)f2bf(f0.x) | ((unsigned)f2bf(f0.y) << 16);
  o.y = (unsigned)f2bf(f0.z) | ((unsigned)f2bf(f0.w) << 16);
  o.z = (unsigned)f2bf(f1.x) | ((unsigned)f2bf(f1.y) << 16);
  o.w = (unsigned)f2bf(f1.z) | ((unsigned)f2bf(f1.w) << 16);
  ((uint4*)d)[i] = o;
}

// ---------------------------------------------------------------------------
// O[m][n] = sum_k X[m][k] * W[n][k]   (X: MxK bf16, W: NxK bf16, O: MxN f32)
// 256 threads = 8 waves; block tile 128x128; wave tile 32x64; K-step 32.
// Double-buffered LDS, async global->LDS copies.
// ---------------------------------------------------------------------------
#define TM 128
#define TN 128
#define TK 32
#define LDS_K 40   /* 80B rows: 16B-aligned b128 chunks + conflict-free gathers */

__global__ __launch_bounds__(256)
void gemm_bf16(const unsigned short* __restrict__ X,
               const unsigned short* __restrict__ W,
               float* __restrict__ O, int M, int N, int K)
{
  __shared__ unsigned short sA[2][TM * LDS_K];
  __shared__ unsigned short sW[2][TN * LDS_K];

  const int tid  = threadIdx.x;
  const int lane = tid & 31;
  const int wave = tid >> 5;
  const int wm   = wave & 3;            // 4 wave-rows  (32 rows each)
  const int wn   = wave >> 2;           // 2 wave-cols  (64 cols each)
  const int m0 = blockIdx.y * TM;
  const int n0 = blockIdx.x * TN;
  const int ml   = lane & 15;
  const int half = lane >> 4;

  // staging map: thread -> (row = tid>>1, 16-col half = tid&1)
  const int r  = tid >> 1;
  const int hc = (tid & 1) * 16;
  const unsigned short* gx = X + (size_t)(m0 + r) * K + hc;
  const unsigned short* gw = W + (size_t)(n0 + r) * K + hc;
  unsigned short* lA = &sA[0][0] + r * LDS_K + hc;
  unsigned short* lW = &sW[0][0] + r * LDS_K + hc;
  const int bstride = TM * LDS_K;

  v8f acc[2][4];
  #pragma unroll
  for (int i = 0; i < 2; ++i)
    #pragma unroll
    for (int j = 0; j < 4; ++j)
      #pragma unroll
      for (int e = 0; e < 8; ++e) acc[i][j][e] = 0.0f;

  // prologue: tile 0 -> buffer 0
  cp16(lA,     gx);  cp16(lA + 8, gx + 8);
  cp16(lW,     gw);  cp16(lW + 8, gw + 8);
  cp_wait();
  __syncthreads();

  int buf = 0;
  for (int k0 = 0; k0 < K; k0 += TK) {
    const int nbuf = buf ^ 1;
    if (k0 + TK < K) {                    // stage next tile into other buffer
      const unsigned short* nx = gx + k0 + TK;
      const unsigned short* nw = gw + k0 + TK;
      cp16(lA + nbuf * bstride,     nx);
      cp16(lA + nbuf * bstride + 8, nx + 8);
      cp16(lW + nbuf * bstride,     nw);
      cp16(lW + nbuf * bstride + 8, nw + 8);
      if (k0 + 2 * TK < K) {              // global_prefetch_b8 one tile ahead
        __builtin_prefetch(gx + k0 + 2 * TK, 0, 0);
        __builtin_prefetch(gw + k0 + 2 * TK, 0, 0);
      }
    }

    const unsigned short* bA = &sA[buf][0];
    const unsigned short* bW = &sW[buf][0];

    // A fragments (2 x 16x32 bf16, ISA lane layout)
    FragAB a[2];
    #pragma unroll
    for (int ms = 0; ms < 2; ++ms) {
      const int row = wm * 32 + ms * 16 + ml;
      #pragma unroll
      for (int v = 0; v < 8; ++v) {
        int kk = (v < 4) ? (half * 8 + 2 * v) : (16 + half * 8 + 2 * (v - 4));
        a[ms].u[v] = *(const unsigned int*)&bA[row * LDS_K + kk];
      }
    }
    // B fragments (4 x 32x16 bf16), each reused by both A fragments
    #pragma unroll
    for (int nt = 0; nt < 4; ++nt) {
      FragAB b;
      const int col = wn * 64 + nt * 16 + ml;
      #pragma unroll
      for (int v = 0; v < 8; ++v) {
        int kk = (v < 4) ? (half * 8 + 2 * v) : (16 + half * 8 + 2 * (v - 4));
        b.u[v] = *(const unsigned int*)&bW[col * LDS_K + kk];
      }
      acc[0][nt] = __builtin_amdgcn_wmma_f32_16x16x32_bf16(
          false, a[0].v, false, b.v, (short)0, acc[0][nt], false, false);
      acc[1][nt] = __builtin_amdgcn_wmma_f32_16x16x32_bf16(
          false, a[1].v, false, b.v, (short)0, acc[1][nt], false, false);
    }
    cp_wait();
    __syncthreads();
    buf = nbuf;
  }

  // epilogue: c[v] -> (M = 8*half + v, N = lane%16)
  #pragma unroll
  for (int ms = 0; ms < 2; ++ms)
    #pragma unroll
    for (int nt = 0; nt < 4; ++nt) {
      FragC c; c.v = acc[ms][nt];
      #pragma unroll
      for (int v = 0; v < 8; ++v) {
        int row = m0 + wm * 32 + ms * 16 + half * 8 + v;
        int col = n0 + wn * 64 + nt * 16 + ml;
        O[(size_t)row * N + col] = c.f[v];
      }
    }
}

// ---------------------------------------------------------------------------
// Depthwise causal conv(4) + SiLU.  x_in = xz[:, :D_INNER]
// ---------------------------------------------------------------------------
__global__ __launch_bounds__(256)
void conv_silu_kernel(const float* __restrict__ xz,
                      const float* __restrict__ cw,
                      const float* __restrict__ cb,
                      float* __restrict__ xact)
{
  int idx = blockIdx.x * blockDim.x + threadIdx.x;
  if (idx >= NTOK * D_INNER) return;
  int d   = idx % D_INNER;
  int tok = idx / D_INNER;
  int t   = tok % SEQ_T;
  float acc = cb[d];
  #pragma unroll
  for (int j = 0; j < D_CONV; ++j) {
    int dtj = j - (D_CONV - 1);
    if (t + dtj >= 0)
      acc += xz[(size_t)(tok + dtj) * (2 * D_INNER) + d] * cw[d * D_CONV + j];
  }
  xact[idx] = acc / (1.0f + __expf(-acc));       // SiLU
}

// ---------------------------------------------------------------------------
// bcdt[tok][0..32] = x_act[tok] . x_proj_w[n]   (N=33, K=2048)
// ---------------------------------------------------------------------------
__global__ __launch_bounds__(256)
void bcdt_kernel(const float* __restrict__ xact,
                 const float* __restrict__ xpw,
                 float* __restrict__ bcdt)
{
  __shared__ float sx[D_INNER];
  __shared__ float red[256];
  int tok = blockIdx.x;
  int tid = threadIdx.x;
  #pragma unroll
  for (int j = 0; j < D_INNER / 256; ++j)
    sx[tid + j * 256] = xact[(size_t)tok * D_INNER + tid + j * 256];
  __syncthreads();
  for (int n = 0; n < NBCDT; ++n) {
    float p = 0.f;
    for (int k = tid; k < D_INNER; k += 256)
      p += sx[k] * xpw[(size_t)n * D_INNER + k];
    red[tid] = p;
    __syncthreads();
    for (int s = 128; s > 0; s >>= 1) {
      if (tid < s) red[tid] += red[tid + s];
      __syncthreads();
    }
    if (tid == 0) bcdt[(size_t)tok * NBCDT + n] = red[0];
    __syncthreads();
  }
}

// ---------------------------------------------------------------------------
// Sequential SSM scan.  Thread = (b, d, s); 16-lane shfl_xor butterfly for
// y = sum_s h*C; fused SiLU(z) gate; writes u directly in bf16 for GEMM2.
// ---------------------------------------------------------------------------
__global__ __launch_bounds__(256)
void scan_kernel(const float* __restrict__ xact,
                 const float* __restrict__ bcdt,
                 const float* __restrict__ xz,       // z gate = cols [D_INNER, 2*D_INNER)
                 const float* __restrict__ dt_w,
                 const float* __restrict__ dt_b,
                 const float* __restrict__ log_A,
                 unsigned short* __restrict__ u)     // bf16 out
{
  const int tid = threadIdx.x;
  const int s   = tid & 15;
  const int d   = blockIdx.x * 16 + (tid >> 4);
  const int b   = blockIdx.y;

  const float A  = -__expf(log_A[d * D_STATE + s]);
  const float dw = dt_w[d];
  const float db = dt_b[d];
  float h = 0.f;
  const size_t tok0 = (size_t)b * SEQ_T;

  for (int t = 0; t < SEQ_T; ++t) {
    const size_t tok = tok0 + t;
    const float* row = bcdt + tok * NBCDT;
    float Bs = row[s];
    float Cs = row[D_STATE + s];
    float ld = row[2 * D_STATE];
    float xa = xact[tok * D_INNER + d];
    float zz = ld * dw + db;
    float dt = (zz > 20.f) ? zz : log1pf(__expf(zz));   // softplus
    h = __expf(A * dt) * h + (xa * dt) * Bs;
    float y = h * Cs;
    #pragma unroll
    for (int off = 1; off < 16; off <<= 1)
      y += __shfl_xor(y, off, 16);
    if (s == 0) {
      float zg = xz[tok * (2 * D_INNER) + D_INNER + d];
      u[tok * D_INNER + d] = f2bf(y * (zg / (1.f + __expf(-zg))));
    }
  }
}

// ---------------------------------------------------------------------------
// out = LayerNorm(o + x) * g + b   (row = token, D_MODEL = 1024)
// ---------------------------------------------------------------------------
__global__ __launch_bounds__(256)
void resid_ln_kernel(const float* __restrict__ o, const float* __restrict__ x,
                     const float* __restrict__ g, const float* __restrict__ be,
                     float* __restrict__ out)
{
  __shared__ float r1[256], r2[256];
  int tok = blockIdx.x, tid = threadIdx.x;
  float vals[D_MODEL / 256];
  float s = 0.f, ss = 0.f;
  #pragma unroll
  for (int j = 0; j < D_MODEL / 256; ++j) {
    int c = tid + j * 256;
    float v = o[(size_t)tok * D_MODEL + c] + x[(size_t)tok * D_MODEL + c];
    vals[j] = v; s += v; ss += v * v;
  }
  r1[tid] = s; r2[tid] = ss;
  __syncthreads();
  for (int st = 128; st > 0; st >>= 1) {
    if (tid < st) { r1[tid] += r1[tid + st]; r2[tid] += r2[tid + st]; }
    __syncthreads();
  }
  float mu  = r1[0] * (1.0f / D_MODEL);
  float var = r2[0] * (1.0f / D_MODEL) - mu * mu;
  float inv = rsqrtf(var + LN_EPS);
  #pragma unroll
  for (int j = 0; j < D_MODEL / 256; ++j) {
    int c = tid + j * 256;
    out[(size_t)tok * D_MODEL + c] = (vals[j] - mu) * inv * g[c] + be[c];
  }
}

// ---------------------------------------------------------------------------
extern "C" void kernel_launch(void* const* d_in, const int* in_sizes, int n_in,
                              void* d_out, int out_size, void* d_ws, size_t ws_size,
                              hipStream_t stream)
{
  const float* x     = (const float*)d_in[0];
  const float* inpw  = (const float*)d_in[1];
  const float* convw = (const float*)d_in[2];
  const float* convb = (const float*)d_in[3];
  const float* xpw   = (const float*)d_in[4];
  const float* dtw   = (const float*)d_in[5];
  const float* dtb   = (const float*)d_in[6];
  const float* logA  = (const float*)d_in[7];
  const float* outpw = (const float*)d_in[8];
  const float* lng   = (const float*)d_in[9];
  const float* lnb   = (const float*)d_in[10];
  float* out = (float*)d_out;

  // ---- workspace layout (fp32 first, then bf16) ----
  float* ws   = (float*)d_ws;
  float* xz   = ws;                 size_t off  = (size_t)NTOK * 2 * D_INNER;
  float* xact = ws + off;                  off += (size_t)NTOK * D_INNER;
  float* bcdt = ws + off;                  off += (size_t)NTOK * NBCDT;
  float* o    = ws + off;                  off += (size_t)NTOK * D_MODEL;
  unsigned short* hb = (unsigned short*)(ws + off);
  unsigned short* xb    = hb;       size_t hoff = (size_t)NTOK * D_MODEL;
  unsigned short* wbin  = hb + hoff;       hoff += (size_t)(2 * D_INNER) * D_MODEL;
  unsigned short* wbout = hb + hoff;       hoff += (size_t)D_MODEL * D_INNER;
  unsigned short* ub    = hb + hoff;

  // 0) bf16 conversions (x, in_proj_w, out_proj_w)
  {
    int n8 = (NTOK * D_MODEL) / 8;
    convert_bf16_kernel<<<(n8 + 255) / 256, 256, 0, stream>>>(x, xb, n8);
    n8 = (2 * D_INNER * D_MODEL) / 8;
    convert_bf16_kernel<<<(n8 + 255) / 256, 256, 0, stream>>>(inpw, wbin, n8);
    n8 = (D_MODEL * D_INNER) / 8;
    convert_bf16_kernel<<<(n8 + 255) / 256, 256, 0, stream>>>(outpw, wbout, n8);
  }
  // 1) xz = x @ in_proj_w^T      (4096x1024 . 1024x4096)
  gemm_bf16<<<dim3((2 * D_INNER) / TN, NTOK / TM), 256, 0, stream>>>(
      xb, wbin, xz, NTOK, 2 * D_INNER, D_MODEL);
  // 2) depthwise conv + SiLU
  conv_silu_kernel<<<(NTOK * D_INNER + 255) / 256, 256, 0, stream>>>(
      xz, convw, convb, xact);
  // 3) bcdt = x_act @ x_proj_w^T  (N=33)
  bcdt_kernel<<<NTOK, 256, 0, stream>>>(xact, xpw, bcdt);
  // 4) SSM scan + SiLU(z) gate -> u (bf16)
  scan_kernel<<<dim3(D_INNER / 16, SEQ_B), 256, 0, stream>>>(
      xact, bcdt, xz, dtw, dtb, logA, ub);
  // 5) o = u @ out_proj_w^T       (4096x2048 . 2048x1024)
  gemm_bf16<<<dim3(D_MODEL / TN, NTOK / TM), 256, 0, stream>>>(
      ub, wbout, o, NTOK, D_MODEL, D_INNER);
  // 6) residual + LayerNorm
  resid_ln_kernel<<<NTOK, 256, 0, stream>>>(o, x, lng, lnb, out);
}
